// DiscreteNarrator_5257039970378
// MI455X (gfx1250) — matrix operand order repaired and checked by reading, not compiled, not run on hardware
//
#include <hip/hip_runtime.h>
#include <math.h>

// ---------------------------------------------------------------------------
// Problem constants
// ---------------------------------------------------------------------------
#define BB   512
#define WW   64
#define DD   512
#define HH   1024
#define NN   1024
#define KK   8
#define CC   64
#define BN   512          // K*C
#define G3   3072         // 3*H
#define COMMIT_W 0.25f

typedef __bf16 bf16_t;
typedef __attribute__((ext_vector_type(16))) __bf16 v16bf;
typedef __attribute__((ext_vector_type(8)))  float  v8f;

// ---------------------------------------------------------------------------
// Workspace layout (bytes, all 256-aligned)
// ---------------------------------------------------------------------------
static constexpr size_t WS_XB     = 0;                                   // bf16 [W*B, D]
static constexpr size_t WS_WIH    = WS_XB     + (size_t)WW*BB*DD*2;      // bf16 [3H, D]
static constexpr size_t WS_WHH    = WS_WIH    + (size_t)G3*DD*2;         // bf16 [3H, H]
static constexpr size_t WS_WQ     = WS_WHH    + (size_t)G3*HH*2;         // bf16 [BN, H]
static constexpr size_t WS_UW1    = WS_WQ     + (size_t)BN*HH*2;         // bf16 [H, 1024]
static constexpr size_t WS_UW2    = WS_UW1    + (size_t)HH*(DD+BN)*2;    // bf16 [H, H]
static constexpr size_t WS_PW1    = WS_UW2    + (size_t)HH*HH*2;         // bf16 [H, BN]
static constexpr size_t WS_PW2    = WS_PW1    + (size_t)HH*BN*2;         // bf16 [BN, H]
static constexpr size_t WS_EMB    = WS_PW2    + (size_t)BN*HH*2;         // bf16 [N, C]
static constexpr size_t WS_GI     = WS_EMB    + (size_t)NN*CC*2;         // f32  [W, B, 3H]
static constexpr size_t WS_H0     = WS_GI     + (size_t)WW*BB*G3*4;      // f32  [B, H]
static constexpr size_t WS_H1     = WS_H0     + (size_t)BB*HH*4;         // f32  [B, H]
static constexpr size_t WS_HBF    = WS_H1     + (size_t)BB*HH*4;         // bf16 [B, H]
static constexpr size_t WS_Q      = WS_HBF    + (size_t)BB*HH*2;         // f32  [B, BN] == [4096, 64]
static constexpr size_t WS_QBF    = WS_Q      + (size_t)BB*BN*4;         // bf16 [4096, 64]
static constexpr size_t WS_DOT    = WS_QBF    + (size_t)BB*BN*2;         // f32  [4096, 1024]
static constexpr size_t WS_E2     = WS_DOT    + (size_t)BB*KK*NN*4;      // f32  [1024] (padded)
static constexpr size_t WS_UINBF  = WS_E2     + 4096;                    // bf16 [B, 1024]
static constexpr size_t WS_HID    = WS_UINBF  + (size_t)BB*(DD+BN)*2;    // f32  [B, H]
static constexpr size_t WS_HIDBF  = WS_HID    + (size_t)BB*HH*4;         // bf16 [B, H]
static constexpr size_t WS_NARRBF = WS_HIDBF  + (size_t)BB*HH*2;         // bf16 [B, BN]

// Output offsets (elements, in return order)
static constexpr size_t O_IDX   = 0;                        // int32 [B,K]
static constexpr size_t O_QUANT = O_IDX   + (size_t)BB*KK;  // f32 [B,K,C]
static constexpr size_t O_NARR  = O_QUANT + (size_t)BB*BN;  // f32 [B,BN]
static constexpr size_t O_UNC   = O_NARR  + (size_t)BB*BN;  // f32 [B,H]
static constexpr size_t O_PRED  = O_UNC   + (size_t)BB*HH;  // f32 [B,BN]
static constexpr size_t O_LOSS  = O_PRED  + (size_t)BB*BN;  // f32 scalar
static constexpr size_t O_LH    = O_LOSS  + 1;              // f32 [B,H]

// ---------------------------------------------------------------------------
// Fragment load helpers
// ---------------------------------------------------------------------------
// A fragment (16-bit A 16x32 layout): lanes 0-15 K={0..7,16..23}, lanes 16-31 K={8..15,24..31}
__device__ __forceinline__ v16bf load_a_frag(const bf16_t* __restrict__ arow, int hi) {
  v16bf a;
  reinterpret_cast<uint4*>(&a)[0] = *reinterpret_cast<const uint4*>(arow + hi * 8);
  reinterpret_cast<uint4*>(&a)[1] = *reinterpret_cast<const uint4*>(arow + 16 + hi * 8);
  return a;
}
// B fragment (32x16): lanes 0-15 col n K={0..7,16..23}, lanes 16-31 K={8..15,24..31}
__device__ __forceinline__ v16bf load_b_frag(const bf16_t* __restrict__ brow, int hi) {
  v16bf b;
  const bf16_t* bp = brow + hi * 8;
  reinterpret_cast<uint4*>(&b)[0] = *reinterpret_cast<const uint4*>(bp);
  reinterpret_cast<uint4*>(&b)[1] = *reinterpret_cast<const uint4*>(bp + 16);
  return b;
}

// ---------------------------------------------------------------------------
// Elementwise helpers
// ---------------------------------------------------------------------------
__global__ void k_cvt_bf16(const float* __restrict__ in, bf16_t* __restrict__ out, int n) {
  for (int i = blockIdx.x * blockDim.x + threadIdx.x; i < n; i += gridDim.x * blockDim.x)
    out[i] = (bf16_t)in[i];
}

__global__ void k_zero_f32(float* __restrict__ p, int n) {
  for (int i = blockIdx.x * blockDim.x + threadIdx.x; i < n; i += gridDim.x * blockDim.x)
    p[i] = 0.0f;
}

// state_window [B,W,D] -> bf16 [(w*B+b), D]
__global__ void k_permute_x(const float* __restrict__ x, bf16_t* __restrict__ out) {
  const int total = BB * WW * DD;
  for (int i = blockIdx.x * blockDim.x + threadIdx.x; i < total; i += gridDim.x * blockDim.x) {
    int d = i % DD;
    int r = i / DD;
    int b = r % BB;
    int w = r / BB;
    out[i] = (bf16_t)x[(size_t)b * WW * DD + (size_t)w * DD + d];
  }
}

// build u_in bf16 [B, D+BN] = concat(state_window[:, -1, :], quantized) ; narr bf16
__global__ void k_build_uin(const float* __restrict__ x, const float* __restrict__ quant,
                            bf16_t* __restrict__ uin, bf16_t* __restrict__ narr) {
  const int total = BB * (DD + BN);
  for (int i = blockIdx.x * blockDim.x + threadIdx.x; i < total; i += gridDim.x * blockDim.x) {
    int c = i % (DD + BN);
    int b = i / (DD + BN);
    float v;
    if (c < DD) {
      v = x[(size_t)b * WW * DD + (size_t)(WW - 1) * DD + c];
    } else {
      v = quant[(size_t)b * BN + (c - DD)];
      narr[(size_t)b * BN + (c - DD)] = (bf16_t)v;
    }
    uin[i] = (bf16_t)v;
  }
}

__global__ void k_emb_norms(const float* __restrict__ emb, float* __restrict__ e2) {
  int n = blockIdx.x * blockDim.x + threadIdx.x;
  if (n < NN) {
    float s = 0.f;
    for (int c = 0; c < CC; ++c) { float v = emb[n * CC + c]; s += v * v; }
    e2[n] = s;
  }
}

// ---------------------------------------------------------------------------
// Generic bf16 WMMA GEMM:  C[M,N] = act(A[M,K] @ Wr[N,K]^T + bias)
// Block = 256 threads (8 waves), block tile 64(M) x 256(N).
// Wave tile 32(M) x 64(N): 2 A-frags x 4 B-frags -> 8 WMMAs per K-chunk.
// act: 0 = identity, 1 = SiLU
// ---------------------------------------------------------------------------
__global__ __launch_bounds__(256)
void k_wmma_gemm(const bf16_t* __restrict__ A, int lda,
                 const bf16_t* __restrict__ Wr, int ldw,
                 const float* __restrict__ bias,
                 float* __restrict__ Cf, bf16_t* __restrict__ Cbf, int ldc,
                 int M, int N, int Ksz, int act) {
  const int lane = threadIdx.x & 31;
  const int wid  = threadIdx.x >> 5;
  const int wm   = wid & 1;           // 2 waves along M
  const int wn   = wid >> 1;          // 4 waves along N
  const int tileM = blockIdx.x * 64 + wm * 32;
  const int tileN = blockIdx.y * 256 + wn * 64;
  if (tileM >= M || tileN >= N) return;

  const int l15 = lane & 15;
  const int hi  = lane >> 4;

  v8f acc[2][4];
#pragma unroll
  for (int i = 0; i < 2; ++i)
#pragma unroll
    for (int t = 0; t < 4; ++t) acc[i][t] = (v8f){0.f,0.f,0.f,0.f,0.f,0.f,0.f,0.f};

  const bf16_t* arow0 = A + (size_t)(tileM + l15) * lda;
  const bf16_t* arow1 = A + (size_t)(tileM + 16 + l15) * lda;
  const bf16_t* brow0 = Wr + (size_t)(tileN + 0  + l15) * ldw;
  const bf16_t* brow1 = Wr + (size_t)(tileN + 16 + l15) * ldw;
  const bf16_t* brow2 = Wr + (size_t)(tileN + 32 + l15) * ldw;
  const bf16_t* brow3 = Wr + (size_t)(tileN + 48 + l15) * ldw;

  for (int k0 = 0; k0 < Ksz; k0 += 32) {
    v16bf a0 = load_a_frag(arow0 + k0, hi);
    v16bf a1 = load_a_frag(arow1 + k0, hi);
    v16bf b0 = load_b_frag(brow0 + k0, hi);
    v16bf b1 = load_b_frag(brow1 + k0, hi);
    v16bf b2 = load_b_frag(brow2 + k0, hi);
    v16bf b3 = load_b_frag(brow3 + k0, hi);

    if (k0 + 32 < Ksz) {            // gfx1250 global_prefetch_b8
      __builtin_prefetch(arow0 + k0 + 32, 0, 0);
      __builtin_prefetch(brow0 + k0 + 32, 0, 0);
    }

    acc[0][0] = __builtin_amdgcn_wmma_f32_16x16x32_bf16(false, a0, false, b0, (short)0, acc[0][0], false, false);
    acc[0][1] = __builtin_amdgcn_wmma_f32_16x16x32_bf16(false, a0, false, b1, (short)0, acc[0][1], false, false);
    acc[0][2] = __builtin_amdgcn_wmma_f32_16x16x32_bf16(false, a0, false, b2, (short)0, acc[0][2], false, false);
    acc[0][3] = __builtin_amdgcn_wmma_f32_16x16x32_bf16(false, a0, false, b3, (short)0, acc[0][3], false, false);
    acc[1][0] = __builtin_amdgcn_wmma_f32_16x16x32_bf16(false, a1, false, b0, (short)0, acc[1][0], false, false);
    acc[1][1] = __builtin_amdgcn_wmma_f32_16x16x32_bf16(false, a1, false, b1, (short)0, acc[1][1], false, false);
    acc[1][2] = __builtin_amdgcn_wmma_f32_16x16x32_bf16(false, a1, false, b2, (short)0, acc[1][2], false, false);
    acc[1][3] = __builtin_amdgcn_wmma_f32_16x16x32_bf16(false, a1, false, b3, (short)0, acc[1][3], false, false);
  }

  // C/D layout: lane 0-15 -> N=lane, VGPR j -> M=j ; lane 16-31 -> N=lane-16, M=j+8
#pragma unroll
  for (int t = 0; t < 4; ++t) {
    const int col = tileN + t * 16 + l15;
    const float bv = bias ? bias[col] : 0.f;
#pragma unroll
    for (int i = 0; i < 2; ++i) {
#pragma unroll
      for (int j = 0; j < 8; ++j) {
        const int row = tileM + i * 16 + j + hi * 8;
        float v = acc[i][t][j] + bv;
        if (act == 1) v = v / (1.f + __expf(-v));   // SiLU
        if (Cf)  Cf [(size_t)row * ldc + col] = v;
        if (Cbf) Cbf[(size_t)row * ldc + col] = (bf16_t)v;
      }
    }
  }
}

// ---------------------------------------------------------------------------
// Fused GRU step: one wave computes a 16x32 tile of h_new (2 N-tiles x 3 gates).
// The f32->bf16 conversion of the h_prev A-fragment is amortized over 6 WMMAs.
// Block = 256 threads = 8 waves -> 16 rows x 256 cols per block.
// grid (B/16=32, H/256=4)
// ---------------------------------------------------------------------------
__global__ __launch_bounds__(256)
void k_gru_step(const float* __restrict__ h_prev, const bf16_t* __restrict__ Whh,
                const float* __restrict__ gi,      // [B, 3H] for this timestep
                const float* __restrict__ b_hh,
                float* __restrict__ h_out) {
  const int lane = threadIdx.x & 31;
  const int wid  = threadIdx.x >> 5;
  const int tileM = blockIdx.x * 16;
  const int tileN = blockIdx.y * 256 + wid * 32;
  const int l15 = lane & 15;
  const int hi  = lane >> 4;

  v8f ar[2], az[2], an[2];
#pragma unroll
  for (int t = 0; t < 2; ++t) {
    ar[t] = (v8f){0.f,0.f,0.f,0.f,0.f,0.f,0.f,0.f};
    az[t] = ar[t];
    an[t] = ar[t];
  }

  const float* hrow = h_prev + (size_t)(tileM + l15) * HH;
  const bf16_t* wr_r[2], * wr_z[2], * wr_n[2];
#pragma unroll
  for (int t = 0; t < 2; ++t) {
    const int n = tileN + t * 16 + l15;
    wr_r[t] = Whh + (size_t)n * HH;
    wr_z[t] = Whh + (size_t)(HH + n) * HH;
    wr_n[t] = Whh + (size_t)(2 * HH + n) * HH;
  }

  for (int k0 = 0; k0 < HH; k0 += 32) {
    // A fragment from f32 h_prev, converted to bf16 in-register
    v16bf a;
    {
      const float* hp = hrow + k0;
      float4 f0 = *reinterpret_cast<const float4*>(hp + hi * 8);
      float4 f1 = *reinterpret_cast<const float4*>(hp + hi * 8 + 4);
      float4 f2 = *reinterpret_cast<const float4*>(hp + 16 + hi * 8);
      float4 f3 = *reinterpret_cast<const float4*>(hp + 16 + hi * 8 + 4);
      a[0]=(bf16_t)f0.x; a[1]=(bf16_t)f0.y; a[2]=(bf16_t)f0.z; a[3]=(bf16_t)f0.w;
      a[4]=(bf16_t)f1.x; a[5]=(bf16_t)f1.y; a[6]=(bf16_t)f1.z; a[7]=(bf16_t)f1.w;
      a[8]=(bf16_t)f2.x; a[9]=(bf16_t)f2.y; a[10]=(bf16_t)f2.z; a[11]=(bf16_t)f2.w;
      a[12]=(bf16_t)f3.x; a[13]=(bf16_t)f3.y; a[14]=(bf16_t)f3.z; a[15]=(bf16_t)f3.w;
    }
#pragma unroll
    for (int t = 0; t < 2; ++t) {
      v16bf br = load_b_frag(wr_r[t] + k0, hi);
      v16bf bz = load_b_frag(wr_z[t] + k0, hi);
      v16bf bn = load_b_frag(wr_n[t] + k0, hi);
      ar[t] = __builtin_amdgcn_wmma_f32_16x16x32_bf16(false, a, false, br, (short)0, ar[t], false, false);
      az[t] = __builtin_amdgcn_wmma_f32_16x16x32_bf16(false, a, false, bz, (short)0, az[t], false, false);
      an[t] = __builtin_amdgcn_wmma_f32_16x16x32_bf16(false, a, false, bn, (short)0, an[t], false, false);
    }
  }

#pragma unroll
  for (int t = 0; t < 2; ++t) {
    const int col = tileN + t * 16 + l15;
    const float bhr = b_hh[col], bhz = b_hh[HH + col], bhn = b_hh[2 * HH + col];
#pragma unroll
    for (int j = 0; j < 8; ++j) {
      const int row = tileM + j + hi * 8;
      const float* girow = gi + (size_t)row * G3;
      float gr = girow[col], gz = girow[HH + col], gn = girow[2 * HH + col];
      float r = 1.f / (1.f + __expf(-(gr + ar[t][j] + bhr)));
      float z = 1.f / (1.f + __expf(-(gz + az[t][j] + bhz)));
      float nn = tanhf(gn + r * (an[t][j] + bhn));
      float hp = h_prev[(size_t)row * HH + col];
      h_out[(size_t)row * HH + col] = (1.f - z) * nn + z * hp;
    }
  }
}

// ---------------------------------------------------------------------------
// Argmin over codes: one wave per (b,k) row. d = |q|^2 + |e|^2 - 2 q.e
// ---------------------------------------------------------------------------
__global__ __launch_bounds__(256)
void k_argmin(const float* __restrict__ queries,   // [4096, 64]
              const float* __restrict__ dot,       // [4096, 1024]
              const float* __restrict__ e2,        // [1024]
              const float* __restrict__ emb,       // [1024, 64]
              int* __restrict__ idx_out,           // [4096]
              float* __restrict__ quant_out) {     // [4096, 64]
  const int lane = threadIdx.x & 31;
  const int row  = blockIdx.x * 8 + (threadIdx.x >> 5);
  if (row >= BB * KK) return;

  float q0 = queries[row * CC + lane];
  float q1 = queries[row * CC + lane + 32];
  float q2 = q0 * q0 + q1 * q1;
#pragma unroll
  for (int off = 16; off > 0; off >>= 1) q2 += __shfl_xor(q2, off, 32);

  float best = 3.4e38f; int bidx = 0;
  for (int n = lane; n < NN; n += 32) {
    float d = q2 + e2[n] - 2.f * dot[(size_t)row * NN + n];
    if (d < best) { best = d; bidx = n; }
  }
#pragma unroll
  for (int off = 16; off > 0; off >>= 1) {
    float ob = __shfl_xor(best, off, 32);
    int   oi = __shfl_xor(bidx, off, 32);
    if (ob < best || (ob == best && oi < bidx)) { best = ob; bidx = oi; }
  }
  if (lane == 0) idx_out[row] = bidx;
  quant_out[row * CC + lane]      = emb[bidx * CC + lane];
  quant_out[row * CC + lane + 32] = emb[bidx * CC + lane + 32];
}

// vq_loss = (1 + COMMIT_W) * mean((q - e)^2)
__global__ __launch_bounds__(256)
void k_vq_loss(const float* __restrict__ q, const float* __restrict__ e,
               float* __restrict__ loss) {
  __shared__ float sm[256];
  const int total = BB * KK * CC;
  float s = 0.f;
  for (int i = blockIdx.x * blockDim.x + threadIdx.x; i < total; i += gridDim.x * blockDim.x) {
    float d = q[i] - e[i];
    s += d * d;
  }
  sm[threadIdx.x] = s;
  __syncthreads();
  for (int st = 128; st > 0; st >>= 1) {
    if (threadIdx.x < st) sm[threadIdx.x] += sm[threadIdx.x + st];
    __syncthreads();
  }
  if (threadIdx.x == 0)
    atomicAdd(loss, sm[0] * ((1.f + COMMIT_W) / (float)total));
}

// ---------------------------------------------------------------------------
// Host launch
// ---------------------------------------------------------------------------
extern "C" void kernel_launch(void* const* d_in, const int* in_sizes, int n_in,
                              void* d_out, int out_size, void* d_ws, size_t ws_size,
                              hipStream_t stream) {
  const float* x    = (const float*)d_in[0];
  const float* Wih  = (const float*)d_in[1];
  const float* Whh  = (const float*)d_in[2];
  const float* bih  = (const float*)d_in[3];
  const float* bhh  = (const float*)d_in[4];
  const float* Wq   = (const float*)d_in[5];
  const float* bq   = (const float*)d_in[6];
  const float* emb  = (const float*)d_in[7];
  const float* uW1  = (const float*)d_in[8];
  const float* ub1  = (const float*)d_in[9];
  const float* uW2  = (const float*)d_in[10];
  const float* ub2  = (const float*)d_in[11];
  const float* pW1  = (const float*)d_in[12];
  const float* pb1  = (const float*)d_in[13];
  const float* pW2  = (const float*)d_in[14];
  const float* pb2  = (const float*)d_in[15];

  char* ws = (char*)d_ws;
  bf16_t* wsXb   = (bf16_t*)(ws + WS_XB);
  bf16_t* wsWih  = (bf16_t*)(ws + WS_WIH);
  bf16_t* wsWhh  = (bf16_t*)(ws + WS_WHH);
  bf16_t* wsWq   = (bf16_t*)(ws + WS_WQ);
  bf16_t* wsUW1  = (bf16_t*)(ws + WS_UW1);
  bf16_t* wsUW2  = (bf16_t*)(ws + WS_UW2);
  bf16_t* wsPW1  = (bf16_t*)(ws + WS_PW1);
  bf16_t* wsPW2  = (bf16_t*)(ws + WS_PW2);
  bf16_t* wsEmb  = (bf16_t*)(ws + WS_EMB);
  float*  wsGi   = (float*)(ws + WS_GI);
  float*  wsH0   = (float*)(ws + WS_H0);
  float*  wsH1   = (float*)(ws + WS_H1);
  bf16_t* wsHbf  = (bf16_t*)(ws + WS_HBF);
  float*  wsQ    = (float*)(ws + WS_Q);
  bf16_t* wsQbf  = (bf16_t*)(ws + WS_QBF);
  float*  wsDot  = (float*)(ws + WS_DOT);
  float*  wsE2   = (float*)(ws + WS_E2);
  bf16_t* wsUin  = (bf16_t*)(ws + WS_UINBF);
  float*  wsHid  = (float*)(ws + WS_HID);
  bf16_t* wsHidB = (bf16_t*)(ws + WS_HIDBF);
  bf16_t* wsNarr = (bf16_t*)(ws + WS_NARRBF);

  float* out     = (float*)d_out;
  int*   outIdx  = (int*)d_out;                // [B,K] at O_IDX
  float* outQ    = out + O_QUANT;
  float* outNarr = out + O_NARR;
  float* outUnc  = out + O_UNC;
  float* outPred = out + O_PRED;
  float* outLoss = out + O_LOSS;
  float* outLH   = out + O_LH;

  const int T = 256;
  // 1) bf16 weight conversions
  k_cvt_bf16<<<dim3((G3*DD+T-1)/T), T, 0, stream>>>(Wih, wsWih, G3*DD);
  k_cvt_bf16<<<dim3((G3*HH+T-1)/T), T, 0, stream>>>(Whh, wsWhh, G3*HH);
  k_cvt_bf16<<<dim3((BN*HH+T-1)/T), T, 0, stream>>>(Wq,  wsWq,  BN*HH);
  k_cvt_bf16<<<dim3((HH*(DD+BN)+T-1)/T), T, 0, stream>>>(uW1, wsUW1, HH*(DD+BN));
  k_cvt_bf16<<<dim3((HH*HH+T-1)/T), T, 0, stream>>>(uW2, wsUW2, HH*HH);
  k_cvt_bf16<<<dim3((HH*BN+T-1)/T), T, 0, stream>>>(pW1, wsPW1, HH*BN);
  k_cvt_bf16<<<dim3((BN*HH+T-1)/T), T, 0, stream>>>(pW2, wsPW2, BN*HH);
  k_cvt_bf16<<<dim3((NN*CC+T-1)/T), T, 0, stream>>>(emb, wsEmb, NN*CC);
  k_permute_x<<<dim3(4096), T, 0, stream>>>(x, wsXb);
  k_zero_f32<<<dim3((BB*HH+T-1)/T), T, 0, stream>>>(wsH0, BB*HH);

  // 2) gi_all = X @ Wih^T + b_ih   [32768, 3072]
  k_wmma_gemm<<<dim3((WW*BB)/64, G3/256), T, 0, stream>>>(
      wsXb, DD, wsWih, DD, bih, wsGi, nullptr, G3, WW*BB, G3, DD, 0);

  // 3) 64 sequential fused GRU steps
  float* hp = wsH0; float* hn = wsH1;
  for (int w = 0; w < WW; ++w) {
    k_gru_step<<<dim3(BB/16, HH/256), T, 0, stream>>>(
        hp, wsWhh, wsGi + (size_t)w * BB * G3, bhh, hn);
    float* t = hp; hp = hn; hn = t;
  }
  // hp now holds last_hidden
  hipMemcpyAsync(outLH, hp, (size_t)BB*HH*4, hipMemcpyDeviceToDevice, stream);
  k_cvt_bf16<<<dim3((BB*HH+T-1)/T), T, 0, stream>>>(hp, wsHbf, BB*HH);

  // 4) queries = h @ Wq^T + bq  [512, 512]  (also emit bf16 view [4096, 64])
  k_wmma_gemm<<<dim3(BB/64, BN/256), T, 0, stream>>>(
      wsHbf, HH, wsWq, HH, bq, wsQ, wsQbf, BN, BB, BN, HH, 0);

  // 5) VQ: dot = q @ emb^T  [4096, 1024], then argmin
  k_emb_norms<<<dim3(4), T, 0, stream>>>(emb, wsE2);
  k_wmma_gemm<<<dim3((BB*KK)/64, NN/256), T, 0, stream>>>(
      wsQbf, CC, wsEmb, CC, nullptr, wsDot, nullptr, NN, BB*KK, NN, CC, 0);
  k_argmin<<<dim3((BB*KK)/8), T, 0, stream>>>(wsQ, wsDot, wsE2, emb,
                                              outIdx + O_IDX, outQ);

  // 6) vq_loss
  k_zero_f32<<<dim3(1), 1, 0, stream>>>(outLoss, 1);
  k_vq_loss<<<dim3(256), T, 0, stream>>>(wsQ, outQ, outLoss);

  // narrator_state == quantized (straight-through forward value)
  hipMemcpyAsync(outNarr, outQ, (size_t)BB*BN*4, hipMemcpyDeviceToDevice, stream);

  // 7) uncertainty head
  k_build_uin<<<dim3((BB*(DD+BN)+T-1)/T), T, 0, stream>>>(x, outQ, wsUin, wsNarr);
  k_wmma_gemm<<<dim3(BB/64, HH/256), T, 0, stream>>>(
      wsUin, DD+BN, wsUW1, DD+BN, ub1, wsHid, wsHidB, HH, BB, HH, DD+BN, 1);
  k_wmma_gemm<<<dim3(BB/64, HH/256), T, 0, stream>>>(
      wsHidB, HH, wsUW2, HH, ub2, outUnc, nullptr, HH, BB, HH, HH, 0);

  // 8) prediction head
  k_wmma_gemm<<<dim3(BB/64, HH/256), T, 0, stream>>>(
      wsNarr, BN, wsPW1, BN, pb1, wsHid, wsHidB, HH, BB, HH, BN, 1);
  k_wmma_gemm<<<dim3(BB/64, BN/256), T, 0, stream>>>(
      wsHidB, HH, wsPW2, HH, pb2, outPred, nullptr, BN, BB, BN, HH, 0);
}
